// InteractionRouter_52544629899287
// MI455X (gfx1250) — matrix operand order repaired and checked by reading, not compiled, not run on hardware
//
#include <hip/hip_runtime.h>
#include <stdint.h>

#define Bb 4
#define Tt 8192
#define Cc 1024
#define Ee 64
#define BT (Bb*Tt)      // 32768
#define KTOP 2

typedef __attribute__((ext_vector_type(16))) __bf16 v16bf;
typedef __attribute__((ext_vector_type(8)))  float  v8f;

union BF16x16 { v16bf v; uint32_t u[8]; uint4 q[2]; };

// pack two f32 -> packed bf16 pair (round-half-up) with a single v_perm_b32:
// D = { hi16(b+0x8000), hi16(a+0x8000) }
static __device__ __forceinline__ uint32_t pack_bf16(float a, float b) {
  uint32_t ua = __float_as_uint(a) + 0x8000u;
  uint32_t ub = __float_as_uint(b) + 0x8000u;
  return __builtin_amdgcn_perm(ub, ua, 0x07060302u);  // bytes: ub3 ub2 ua3 ua2
}
static __device__ __forceinline__ uint16_t f2bf_rne(float f) {
  uint32_t u = __float_as_uint(f);
  uint32_t r = u + 0x7FFFu + ((u >> 16) & 1u);
  return (uint16_t)(r >> 16);
}

// xor-lane exchange via ds_swizzle BITMODE: new_lane = (lane & 0x1f) ^ M  (one ds op)
template<int M> static __device__ __forceinline__ float swz_f(float v) {
  return __int_as_float(__builtin_amdgcn_ds_swizzle(__float_as_int(v), 0x1f | (M << 10)));
}
template<int M> static __device__ __forceinline__ int swz_i(int v) {
  return __builtin_amdgcn_ds_swizzle(v, 0x1f | (M << 10));
}
// merge two sorted top-2 pairs across xor-lane M (lower index wins ties)
template<int M> static __device__ __forceinline__ void merge2(float& v1, int& e1,
                                                              float& v2, int& e2) {
  float ov1 = swz_f<M>(v1), ov2 = swz_f<M>(v2);
  int   oe1 = swz_i<M>(e1), oe2 = swz_i<M>(e2);
  bool otherFirst = (ov1 > v1) || (ov1 == v1 && oe1 < e1);
  if (otherFirst) {
    bool mineSecond = (v1 > ov2) || (v1 == ov2 && e1 < oe2);
    v2 = mineSecond ? v1 : ov2;  e2 = mineSecond ? e1 : oe2;
    v1 = ov1; e1 = oe1;
  } else {
    if ((ov1 > v2) || (ov1 == v2 && oe1 < e2)) { v2 = ov1; e2 = oe1; }
  }
}

// hcond[b,c] = sum_d cond[b,d] * Wc[c, 1024+d]
__global__ void __launch_bounds__(256) k_hcond(const float* __restrict__ cond,
                                               const float* __restrict__ Wc,
                                               float* __restrict__ hcond) {
  int gid = blockIdx.x * 256 + threadIdx.x;           // 0..4095
  int b = gid >> 10, c = gid & 1023;
  const float* cv = cond + b * Cc;
  const float* wr = Wc + (size_t)c * (2 * Cc) + Cc;
  float s = 0.f;
  #pragma unroll 4
  for (int d = 0; d < Cc; ++d) s = fmaf(cv[d], wr[d], s);
  hcond[gid] = s;
}

// bias[b,e] = sum_c hcond[b,c] * Wg[e,c]
__global__ void __launch_bounds__(256) k_bias(const float* __restrict__ hcond,
                                              const float* __restrict__ Wg,
                                              float* __restrict__ bias) {
  int tid = threadIdx.x;                              // 0..255
  int b = tid >> 6, e = tid & 63;
  const float* h = hcond + b * Cc;
  const float* w = Wg + e * Cc;
  float s = 0.f;
  #pragma unroll 4
  for (int c = 0; c < Cc; ++c) s = fmaf(h[c], w[c], s);
  bias[tid] = s;
}

// Mx[e,d] = sum_c Wg[e,c] * Wc[c,d]  (d < 1024), stored bf16 row-major
__global__ void __launch_bounds__(256) k_mx(const float* __restrict__ Wg,
                                            const float* __restrict__ Wc,
                                            uint16_t* __restrict__ mx) {
  int gid = blockIdx.x * 256 + threadIdx.x;           // 0..65535
  int e = gid >> 10, d = gid & 1023;
  const float* wg = Wg + e * Cc;
  float s = 0.f;
  #pragma unroll 4
  for (int c = 0; c < Cc; ++c) s = fmaf(wg[c], Wc[(size_t)c * (2 * Cc) + d], s);
  mx[gid] = f2bf_rne(s);
}

// Main streaming kernel: logits = x @ Mx^T + bias, fused softmax/top-2.
// Block = 128 threads = 4 waves; each wave handles 32 tokens x 64 experts.
#define SM_STRIDE 132                                  // 128 data dwords + 4 pad (bank spread)
__global__ void __launch_bounds__(128) k_main(const float* __restrict__ x,
                                              const uint16_t* __restrict__ mx,
                                              const float* __restrict__ bias,
                                              float* __restrict__ o_idx,
                                              float* __restrict__ o_sc,
                                              float* __restrict__ o_pr) {
  __shared__ uint32_t smem[Ee * SM_STRIDE];            // 33 KB: Mx[:, 256-wide K chunk]
  const int tid  = threadIdx.x;
  const int wave = tid >> 5;
  const int lane = tid & 31;
  const int l15  = lane & 15;
  const int hf   = lane >> 4;
  const int tokenBase = (blockIdx.x * 4 + wave) * 32;
  const int b = tokenBase >> 13;                       // T=8192; 32-token chunk never crosses b

  v8f acc[2][4];
  #pragma unroll
  for (int t = 0; t < 2; ++t)
    #pragma unroll
    for (int nt = 0; nt < 4; ++nt) {
      float be = bias[b * Ee + nt * 16 + l15];         // bias folded into C-init
      v8f a;
      #pragma unroll
      for (int j = 0; j < 8; ++j) a[j] = be;
      acc[t][nt] = a;
    }

  const float4* row0 = (const float4*)(x + (size_t)(tokenBase + l15) * Cc);
  const float4* row1 = (const float4*)(x + (size_t)(tokenBase + 16 + l15) * Cc);
  const uint4* mx4 = (const uint4*)mx;                 // Mx row = 128 uint4

  for (int kh = 0; kh < 4; ++kh) {                     // 4 stages of 256 K
    __syncthreads();
    #pragma unroll
    for (int j = tid; j < Ee * 32; j += 128) {         // 2048 uint4 copies, 16 iters
      int e = j >> 5, c4 = j & 31;
      ((uint4*)smem)[e * (SM_STRIDE / 4) + c4] = mx4[e * 128 + kh * 32 + c4];
    }
    __syncthreads();
    for (int ks = 0; ks < 8; ++ks) {                   // 8 K-steps of 32 per stage
      const int kq = (kh * 256 + ks * 32) >> 2;        // float4 column index
      // Issue all global A loads first...
      float4 f00 = row0[kq + hf * 2 + 0];
      float4 f01 = row0[kq + hf * 2 + 1];
      float4 f02 = row0[kq + hf * 2 + 4];
      float4 f03 = row0[kq + hf * 2 + 5];
      float4 f10 = row1[kq + hf * 2 + 0];
      float4 f11 = row1[kq + hf * 2 + 1];
      float4 f12 = row1[kq + hf * 2 + 4];
      float4 f13 = row1[kq + hf * 2 + 5];
      // ...then all 8 LDS B-fragment loads (single dscnt wait covers all 8 WMMAs)...
      uint4 bq[8];
      #pragma unroll
      for (int nt = 0; nt < 4; ++nt) {
        const int q = (nt * 16 + l15) * (SM_STRIDE / 4) + ks * 4 + hf;
        bq[2 * nt + 0] = ((const uint4*)smem)[q];
        bq[2 * nt + 1] = ((const uint4*)smem)[q + 2];
      }
      // ...conversion VALU overlaps the LDS latency.
      BF16x16 a0, a1;
      a0.u[0] = pack_bf16(f00.x, f00.y); a0.u[1] = pack_bf16(f00.z, f00.w);
      a0.u[2] = pack_bf16(f01.x, f01.y); a0.u[3] = pack_bf16(f01.z, f01.w);
      a0.u[4] = pack_bf16(f02.x, f02.y); a0.u[5] = pack_bf16(f02.z, f02.w);
      a0.u[6] = pack_bf16(f03.x, f03.y); a0.u[7] = pack_bf16(f03.z, f03.w);
      a1.u[0] = pack_bf16(f10.x, f10.y); a1.u[1] = pack_bf16(f10.z, f10.w);
      a1.u[2] = pack_bf16(f11.x, f11.y); a1.u[3] = pack_bf16(f11.z, f11.w);
      a1.u[4] = pack_bf16(f12.x, f12.y); a1.u[5] = pack_bf16(f12.z, f12.w);
      a1.u[6] = pack_bf16(f13.x, f13.y); a1.u[7] = pack_bf16(f13.z, f13.w);
      #pragma unroll
      for (int nt = 0; nt < 4; ++nt) {
        BF16x16 bm; bm.q[0] = bq[2 * nt]; bm.q[1] = bq[2 * nt + 1];
        acc[0][nt] = __builtin_amdgcn_wmma_f32_16x16x32_bf16(
            false, a0.v, false, bm.v, (short)0, acc[0][nt], false, false);
        acc[1][nt] = __builtin_amdgcn_wmma_f32_16x16x32_bf16(
            false, a1.v, false, bm.v, (short)0, acc[1][nt], false, false);
      }
    }
  }

  // Epilogue: each 16-lane half owns 8 token rows; expert e = nt*16 + (lane&15).
  #pragma unroll
  for (int t = 0; t < 2; ++t) {
    #pragma unroll
    for (int r = 0; r < 8; ++r) {
      const int tok = tokenBase + t * 16 + hf * 8 + r;
      float l0 = acc[t][0][r], l1 = acc[t][1][r], l2 = acc[t][2][r], l3 = acc[t][3][r];
      float m = fmaxf(fmaxf(l0, l1), fmaxf(l2, l3));
      m = fmaxf(m, swz_f<1>(m)); m = fmaxf(m, swz_f<2>(m));
      m = fmaxf(m, swz_f<4>(m)); m = fmaxf(m, swz_f<8>(m));
      float p0 = __expf(l0 - m), p1 = __expf(l1 - m), p2 = __expf(l2 - m), p3 = __expf(l3 - m);
      float s = (p0 + p1) + (p2 + p3);
      s += swz_f<1>(s); s += swz_f<2>(s); s += swz_f<4>(s); s += swz_f<8>(s);
      float inv = __builtin_amdgcn_rcpf(s);
      float* pr = o_pr + (size_t)tok * Ee + l15;
      pr[0] = p0 * inv; pr[16] = p1 * inv; pr[32] = p2 * inv; pr[48] = p3 * inv;

      // local top-2 over this lane's 4 logits (prefer lower index on ties)
      float v1 = l0, v2 = -3.4e38f; int e1 = l15, e2 = 0x7fffffff;
      {
        float lv[3] = {l1, l2, l3};
        #pragma unroll
        for (int j = 0; j < 3; ++j) {
          float v = lv[j]; int e = (j + 1) * 16 + l15;
          if (v > v1) { v2 = v1; e2 = e1; v1 = v; e1 = e; }
          else if (v > v2) { v2 = v; e2 = e; }
        }
      }
      merge2<1>(v1, e1, v2, e2);
      merge2<2>(v1, e1, v2, e2);
      merge2<4>(v1, e1, v2, e2);
      merge2<8>(v1, e1, v2, e2);
      if (l15 == 0) {
        o_idx[tok * KTOP + 0] = (float)e1;
        o_idx[tok * KTOP + 1] = (float)e2;
        float s0 = 1.f / (1.f + __expf(v2 - v1));      // softmax over [v1,v2]
        o_sc[tok * KTOP + 0] = s0;
        o_sc[tok * KTOP + 1] = 1.f - s0;
      }
    }
  }
}

// Deterministic importance/load: one block per expert, fixed-order sums + LDS tree.
__global__ void __launch_bounds__(256) k_stats(const float* __restrict__ probs,
                                               const float* __restrict__ idxv,
                                               float* __restrict__ imp,
                                               float* __restrict__ load) {
  __shared__ float sm[256];
  int e = blockIdx.x, tid = threadIdx.x;
  float s = 0.f;
  for (int i = 0; i < BT / 256; ++i) s += probs[(size_t)(i * 256 + tid) * Ee + e];
  sm[tid] = s; __syncthreads();
  for (int st = 128; st > 0; st >>= 1) { if (tid < st) sm[tid] += sm[tid + st]; __syncthreads(); }
  if (tid == 0) imp[e] = sm[0] * (1.0f / BT);
  __syncthreads();
  float c = 0.f;
  for (int i = 0; i < (BT * KTOP) / 256; ++i) {
    float v = idxv[i * 256 + tid];
    c += ((int)(v + 0.5f) == e) ? 1.f : 0.f;
  }
  sm[tid] = c; __syncthreads();
  for (int st = 128; st > 0; st >>= 1) { if (tid < st) sm[tid] += sm[tid + st]; __syncthreads(); }
  if (tid == 0) load[e] = sm[0] * (1.0f / (BT * KTOP));
}

extern "C" void kernel_launch(void* const* d_in, const int* in_sizes, int n_in,
                              void* d_out, int out_size, void* d_ws, size_t ws_size,
                              hipStream_t stream) {
  (void)in_sizes; (void)n_in; (void)out_size; (void)ws_size;
  const float* x    = (const float*)d_in[0];
  const float* cond = (const float*)d_in[1];
  const float* Wg   = (const float*)d_in[2];
  const float* Wc   = (const float*)d_in[3];

  float* out    = (float*)d_out;
  float* o_idx  = out;                                  // BT*2
  float* o_sc   = out + (size_t)BT * KTOP;              // BT*2
  float* o_pr   = out + (size_t)2 * BT * KTOP;          // BT*64
  float* o_imp  = o_pr + (size_t)BT * Ee;               // 64
  float* o_load = o_imp + Ee;                           // 64

  char* ws = (char*)d_ws;
  float*    hcond = (float*)ws;                         // 16 KB
  float*    bias  = (float*)(ws + 16384);               // 1 KB
  uint16_t* mxb   = (uint16_t*)(ws + 17408);            // 128 KB (16B aligned)

  k_hcond<<<dim3(16),  dim3(256), 0, stream>>>(cond, Wc, hcond);
  k_mx   <<<dim3(256), dim3(256), 0, stream>>>(Wg, Wc, mxb);
  k_bias <<<dim3(1),   dim3(256), 0, stream>>>(hcond, Wg, bias);
  k_main <<<dim3(256), dim3(128), 0, stream>>>(x, mxb, bias, o_idx, o_sc, o_pr);
  k_stats<<<dim3(64),  dim3(256), 0, stream>>>(o_pr, o_idx, o_imp, o_load);
}
